// SeqCap_9706626089112
// MI455X (gfx1250) — matrix operand order repaired and checked by reading, not compiled, not run on hardware
//
#include <hip/hip_runtime.h>
#include <hip/hip_bf16.h>
#include <math.h>

typedef _Float16 v16h __attribute__((ext_vector_type(16)));
typedef _Float16 v8h  __attribute__((ext_vector_type(8)));
typedef float    v8f  __attribute__((ext_vector_type(8)));

#define B_       512
#define I_       512
#define DIN_     32
#define J_       10
#define D_       16
#define N_       (J_ * D_)      // 160
#define ROUT_    3
#define NTHREADS 512
#define NWAVES   (NTHREADS / 32)

// One workgroup == one batch element. u (I x J*D) lives in LDS as f16 (160 KB),
// computed with v_wmma_f32_16x16x32_f16 (K = DIN = 32 -> exactly one WMMA per tile).
__global__ __launch_bounds__(NTHREADS)
void capsule_fused_kernel(const float* __restrict__ xg,
                          const float* __restrict__ Wg,
                          float* __restrict__ out)
{
    __shared__ __attribute__((aligned(32))) _Float16 U[I_ * N_];     // 163840 B
    __shared__ __attribute__((aligned(32))) _Float16 Wt[N_ * DIN_];  //  10240 B (W transposed [n][k])
    __shared__ float bb[J_ * I_];                                    //  20480 B routing logits b
    __shared__ float cc[J_ * I_];                                    //  20480 B coupling coeffs c
    __shared__ float oo[J_ * D_];                                    //    640 B output capsules o
    __shared__ float lg[16];                                         // final logits

    const int tid   = threadIdx.x;
    const int lane  = tid & 31;
    const int wave  = tid >> 5;
    const int batch = blockIdx.x;

    // ---------------- stage W (transposed, f16) + zero b ----------------
    for (int idx = tid; idx < DIN_ * N_; idx += NTHREADS) {
        int k = idx / N_;
        int n = idx - k * N_;
        Wt[n * DIN_ + k] = (_Float16)Wg[idx];
    }
    for (int idx = tid; idx < J_ * I_; idx += NTHREADS) bb[idx] = 0.0f;
    __syncthreads();

    // ---------------- u = x @ W via WMMA (f16 in, f32 acc) ----------------
    const float* xb      = xg + (size_t)batch * I_ * DIN_;
    const int    halfsel = lane >> 4;   // 0: lanes 0-15, 1: lanes 16-31
    const int    mlane   = lane & 15;

    for (int mt = wave; mt < I_ / 16; mt += NWAVES) {
        const int row = mt * 16 + mlane;
        // A operand (16x32 f16): lane<16 holds K {0..7,16..23}; lane>=16 holds {8..15,24..31}
        const float* ap = xb + row * DIN_ + halfsel * 8;
        v16h a;
        #pragma unroll
        for (int t = 0; t < 8; ++t) {
            a[t]     = (_Float16)ap[t];
            a[t + 8] = (_Float16)ap[t + 16];
        }
        #pragma unroll
        for (int nt = 0; nt < J_; ++nt) {
            // B operand (32x16 f16): lane -> column n = nt*16+mlane;
            // lane<16 holds K {0..7,16..23}, lane>=16 holds K {8..15,24..31}
            const _Float16* bp = &Wt[(nt * 16 + mlane) * DIN_ + halfsel * 8];
            v8h blo = *(const v8h*)(bp);
            v8h bhi = *(const v8h*)(bp + 16);
            v16h bmat;
            #pragma unroll
            for (int t = 0; t < 8; ++t) { bmat[t] = blo[t]; bmat[t + 8] = bhi[t]; }

            v8f acc = {};
            acc = __builtin_amdgcn_wmma_f32_16x16x32_f16(
                /*neg_a=*/false, a, /*neg_b=*/false, bmat,
                /*c_mod=*/(short)0, acc, /*reuse_a=*/false, /*reuse_b=*/false);

            // D layout: VGPR r -> row (r + 8*halfsel), col = mlane
            #pragma unroll
            for (int r = 0; r < 8; ++r) {
                U[(mt * 16 + r + 8 * halfsel) * N_ + nt * 16 + mlane] = (_Float16)acc[r];
            }
        }
    }
    __syncthreads();

    // ---------------- dynamic routing (3 rounds, all in LDS) ----------------
    for (int rr = 0; rr < ROUT_; ++rr) {
        // c = softmax(b) over j, per column i
        for (int i = tid; i < I_; i += NTHREADS) {
            float bv[J_];
            float m = -1e30f;
            #pragma unroll
            for (int j = 0; j < J_; ++j) { bv[j] = bb[j * I_ + i]; m = fmaxf(m, bv[j]); }
            float s = 0.0f;
            #pragma unroll
            for (int j = 0; j < J_; ++j) { bv[j] = __expf(bv[j] - m); s += bv[j]; }
            float inv = 1.0f / s;
            #pragma unroll
            for (int j = 0; j < J_; ++j) cc[j * I_ + i] = bv[j] * inv;
        }
        for (int idx = tid; idx < J_ * D_; idx += NTHREADS) oo[idx] = 0.0f;
        __syncthreads();

        // o[j][d] = sum_i c[j][i] * U[i][j*16+d]
        // lane -> (d = lane>>1, parity s = lane&1); wave w covers i in [w*32, w*32+32)
        {
            const int d = lane >> 1;
            const int s = lane & 1;
            const int ibase = wave * 32;
            for (int j = 0; j < J_; ++j) {
                float acc = 0.0f;
                #pragma unroll
                for (int t = 0; t < 16; ++t) {
                    const int i = ibase + s + 2 * t;
                    acc += cc[j * I_ + i] * (float)U[i * N_ + j * 16 + d];
                }
                atomicAdd(&oo[j * 16 + d], acc);   // ds_add_f32
            }
        }
        __syncthreads();

        // squash(o) per output capsule j
        if (tid < J_) {
            float sq = 0.0f;
            #pragma unroll
            for (int d2 = 0; d2 < D_; ++d2) { float v = oo[tid * 16 + d2]; sq += v * v; }
            float scale = (sq / (1.0f + sq)) * rsqrtf(sq + 1e-7f);
            #pragma unroll
            for (int d2 = 0; d2 < D_; ++d2) oo[tid * 16 + d2] *= scale;
        }
        __syncthreads();

        if (rr < ROUT_ - 1) {
            // b[j][i] += sum_d o[j][d] * U[i][j*16+d]
            for (int i = tid; i < I_; i += NTHREADS) {
                #pragma unroll
                for (int j = 0; j < J_; ++j) {
                    const _Float16* up = &U[i * N_ + j * 16];
                    v8h u0 = *(const v8h*)(up);
                    v8h u1 = *(const v8h*)(up + 8);
                    float dot = 0.0f;
                    #pragma unroll
                    for (int d2 = 0; d2 < 8; ++d2) {
                        dot += oo[j * 16 + d2]     * (float)u0[d2];
                        dot += oo[j * 16 + d2 + 8] * (float)u1[d2];
                    }
                    bb[j * I_ + i] += dot;
                }
            }
            __syncthreads();
        }
    }

    // ---------------- logits = sum_d o ; softmax over j ----------------
    if (tid < J_) {
        float s = 0.0f;
        #pragma unroll
        for (int d2 = 0; d2 < D_; ++d2) s += oo[tid * 16 + d2];
        lg[tid] = s;
    }
    __syncthreads();
    if (tid == 0) {
        float m = -1e30f;
        for (int j = 0; j < J_; ++j) m = fmaxf(m, lg[j]);
        float e[J_];
        float s = 0.0f;
        for (int j = 0; j < J_; ++j) { e[j] = __expf(lg[j] - m); s += e[j]; }
        float inv = 1.0f / s;
        for (int j = 0; j < J_; ++j) out[(size_t)batch * J_ + j] = e[j] * inv;
    }
}

extern "C" void kernel_launch(void* const* d_in, const int* in_sizes, int n_in,
                              void* d_out, int out_size, void* d_ws, size_t ws_size,
                              hipStream_t stream) {
    (void)in_sizes; (void)n_in; (void)d_ws; (void)ws_size; (void)out_size;
    const float* x = (const float*)d_in[0];   // [512, 512, 32] f32
    const float* W = (const float*)d_in[1];   // [32, 160] f32
    float* out = (float*)d_out;               // [512, 10] f32
    capsule_fused_kernel<<<B_, NTHREADS, 0, stream>>>(x, W, out);
}
